// DynamicMoE_40312563040937
// MI455X (gfx1250) — compile-verified, matrix operand order checked by reading, not run on hardware
//
#include <hip/hip_runtime.h>
#include <hip/hip_bf16.h>

// ---------------------------------------------------------------------------
// MoE (top-1 gating) for MI455X / gfx1250, bf16 WMMA path.
// S=8192 tokens, M=1024, F=4096, E=16 experts, capacity C=512.
// ---------------------------------------------------------------------------

#define S_TOK   8192
#define D_MODEL 1024
#define D_FF    4096
#define N_EXP   16
#define CAP     512

typedef __bf16 v16bf __attribute__((ext_vector_type(16)));
typedef __bf16 v8bf  __attribute__((ext_vector_type(8)));
typedef __bf16 v2bf  __attribute__((ext_vector_type(2)));
typedef float  v8f   __attribute__((ext_vector_type(8)));
typedef float  v4f   __attribute__((ext_vector_type(4)));

// Build a 16x32 bf16 A/B fragment per the CDNA5 WMMA register layout:
// lanes 0-15 take K=[k..k+7] and [k+16..k+23]; lanes 16-31 take [k+8..k+15]
// and [k+24..k+31].  `p` points at the row start + k0; halfOff = (lane>>4)*8.
__device__ __forceinline__ v16bf load_frag16(const __bf16* p, int halfOff) {
    v8bf lo = *(const v8bf*)(p + halfOff);
    v8bf hi = *(const v8bf*)(p + 16 + halfOff);
    return __builtin_shufflevector(lo, hi, 0, 1, 2, 3, 4, 5, 6, 7,
                                           8, 9, 10, 11, 12, 13, 14, 15);
}

__device__ __forceinline__ float fast_tanh(float x) {
#if __has_builtin(__builtin_amdgcn_tanhf)
    return __builtin_amdgcn_tanhf(x);   // v_tanh_f32 on gfx1250
#else
    return tanhf(x);
#endif
}

__device__ __forceinline__ float gelu_tanh(float x) {
    const float k0 = 0.7978845608028654f, k1 = 0.044715f;
    float t = fast_tanh(k0 * (x + k1 * x * x * x));
    return 0.5f * x * (1.0f + t);
}

// ---------------------------------------------------------------------------
// 0) init: zero main output, me_sum accumulators, slot_token = -1
// ---------------------------------------------------------------------------
__global__ void moe_init(float* __restrict__ out, float* __restrict__ me_sum,
                         int* __restrict__ slot_token) {
    size_t i = (size_t)blockIdx.x * 256 + threadIdx.x;
    if (i < (size_t)S_TOK * D_MODEL) out[i] = 0.0f;
    if (i < N_EXP) me_sum[i] = 0.0f;
    if (i < N_EXP * CAP) slot_token[i] = -1;
}

// ---------------------------------------------------------------------------
// 1) gating: one wave per token.  logits = x[s,:] @ wg; softmax; argmax.
// ---------------------------------------------------------------------------
__global__ __launch_bounds__(256) void moe_gate(
        const float* __restrict__ x, const float* __restrict__ wg,
        int* __restrict__ expert_id, float* __restrict__ gate,
        float* __restrict__ me_sum) {
    __shared__ float wgs[D_MODEL * N_EXP];   // 64 KB
    __shared__ float me_loc[N_EXP];
    const int tid = threadIdx.x;
    if (tid < N_EXP) me_loc[tid] = 0.0f;
    for (int i = tid; i < D_MODEL * N_EXP; i += 256) wgs[i] = wg[i];
    __syncthreads();

    const int lane = tid & 31, wid = tid >> 5;
    const int s = blockIdx.x * 8 + wid;   // grid = 1024 -> exactly 8192 tokens
    const float* xr = x + (size_t)s * D_MODEL;

    float lg[N_EXP];
#pragma unroll
    for (int e = 0; e < N_EXP; ++e) lg[e] = 0.0f;

    for (int i = 0; i < D_MODEL / 32; ++i) {
        const int m = lane + 32 * i;
        const float xv = xr[m];
        const v4f* row = (const v4f*)(wgs + m * N_EXP);
        const v4f r0 = row[0], r1 = row[1], r2 = row[2], r3 = row[3];
#pragma unroll
        for (int j = 0; j < 4; ++j) {
            lg[j]      += xv * r0[j];
            lg[4 + j]  += xv * r1[j];
            lg[8 + j]  += xv * r2[j];
            lg[12 + j] += xv * r3[j];
        }
    }
#pragma unroll
    for (int off = 16; off >= 1; off >>= 1)
#pragma unroll
        for (int e = 0; e < N_EXP; ++e) lg[e] += __shfl_xor(lg[e], off, 32);

    if (lane == 0) {
        float mx = lg[0]; int imax = 0;
#pragma unroll
        for (int e = 1; e < N_EXP; ++e)
            if (lg[e] > mx) { mx = lg[e]; imax = e; }
        float p[N_EXP]; float sum = 0.0f;
#pragma unroll
        for (int e = 0; e < N_EXP; ++e) { p[e] = __expf(lg[e] - mx); sum += p[e]; }
        const float inv = 1.0f / sum;
        expert_id[s] = imax;
        gate[s] = p[imax] * inv;
#pragma unroll
        for (int e = 0; e < N_EXP; ++e) atomicAdd(&me_loc[e], p[e] * inv);
    }
    __syncthreads();
    if (tid < N_EXP) atomicAdd(&me_sum[tid], me_loc[tid]);
}

// ---------------------------------------------------------------------------
// 2) scan: 1 block, wave e owns expert e.  Deterministic in-order capacity
//    slot assignment via ballot prefix-sum (== cumsum in the reference).
//    Also writes l_aux and exp_counts at the tail of d_out.
// ---------------------------------------------------------------------------
__global__ __launch_bounds__(512) void moe_scan(
        const int* __restrict__ expert_id, const float* __restrict__ me_sum,
        int* __restrict__ slot_token, float* __restrict__ out_tail) {
    __shared__ int unc[N_EXP];
    const int tid = threadIdx.x, lane = tid & 31, w = tid >> 5;
    int base = 0;
    for (int t0 = 0; t0 < S_TOK; t0 += 32) {
        const int id = expert_id[t0 + lane];
        const bool match = (id == w);
        const unsigned mask = (unsigned)__ballot(match);
        const int pos = base + __popc(mask & ((1u << lane) - 1u));
        if (match && pos < CAP) slot_token[w * CAP + pos] = t0 + lane;
        base += __popc(mask);
    }
    if (lane == 0) unc[w] = base;
    __syncthreads();
    if (tid == 0) {
        float acc = 0.0f;
        for (int e = 0; e < N_EXP; ++e) acc += me_sum[e] * (float)unc[e];
        out_tail[0] = acc * (float)N_EXP / ((float)S_TOK * (float)S_TOK);
    }
    if (tid < N_EXP) {
        int c = unc[tid] < CAP ? unc[tid] : CAP;  // capped counts
        ((int*)(out_tail + 1))[tid] = c;
    }
}

// ---------------------------------------------------------------------------
// 3) pack: gather kept tokens into bf16 dispatch buffer Xd[E*C, M].
//    Packed pair stores -> v_cvt_pk_bf16_f32 + global_store_b32.
// ---------------------------------------------------------------------------
__global__ __launch_bounds__(256) void moe_pack(
        const float* __restrict__ x, const int* __restrict__ slot_token,
        __bf16* __restrict__ Xd) {
    const int row = blockIdx.x;              // e*CAP + c
    const int s = slot_token[row];
    const float2* src = (const float2*)(x + (size_t)(s < 0 ? 0 : s) * D_MODEL);
    v2bf* dst = (v2bf*)(Xd + (size_t)row * D_MODEL);
#pragma unroll
    for (int j = 0; j < 2; ++j) {
        const int p = threadIdx.x + 256 * j;
        float2 v;
        if (s >= 0) v = src[p]; else { v.x = 0.0f; v.y = 0.0f; }
        dst[p] = v2bf{(__bf16)v.x, (__bf16)v.y};
    }
}

// ---------------------------------------------------------------------------
// GEMM core (shared by FFN1/FFN2):
//   block = 256 threads (8 waves), output tile 128(M) x 128(N).
//   wave tile = 32(M) x 64(N): 2 A fragments x 4 B fragments = 8 WMMA / K-step
//   (B fragments amortized over two A rows -> half the LDS reads per FLOP).
//   B tile (32K x 128N, fp32 in HBM) is software-pipelined: fp32 values for
//   step k+1 are fetched into registers while step k's WMMAs run, then
//   converted to bf16 and stored transposed into LDS (padded stride 40:
//   16 B-aligned rows, conflict-free across 16 consecutive rows).
// ---------------------------------------------------------------------------
#define BT_STRIDE 40

__global__ __launch_bounds__(256) void moe_ffn1(
        const float* __restrict__ w1, const float* __restrict__ b1,
        const __bf16* __restrict__ Xd, __bf16* __restrict__ H) {
    const int e = blockIdx.z;
    const int mblk = blockIdx.x * 128;
    const int nblk = blockIdx.y * 128;
    const int tid = threadIdx.x;
    const int lane = tid & 31, wid = tid >> 5;
    const int waveM = wid & 3, waveN = wid >> 2;
    const int halfOff = (lane >> 4) * 8;

    __shared__ __bf16 Bt[128 * BT_STRIDE];

    const float* W = w1 + (size_t)e * D_MODEL * D_FF;
    const __bf16* A0 = Xd + ((size_t)e * CAP + mblk + waveM * 32 + (lane & 15)) * D_MODEL;
    const __bf16* A1 = A0 + (size_t)16 * D_MODEL;

    v8f acc[2][4];
#pragma unroll
    for (int f = 0; f < 2; ++f)
#pragma unroll
        for (int nt = 0; nt < 4; ++nt) acc[f][nt] = v8f{};

    const int cg = tid & 31;   // 4 cols each -> 128 cols
    const int rg = tid >> 5;   // 4 rows each -> 32 rows
    v4f pre[4];

    auto loadTile = [&](int k0) {
#pragma unroll
        for (int r = 0; r < 4; ++r)
            pre[r] = *(const v4f*)(W + (size_t)(k0 + rg * 4 + r) * D_FF + nblk + cg * 4);
    };
    auto storeTile = [&]() {
#pragma unroll
        for (int r = 0; r < 4; ++r) {
            const int kk = rg * 4 + r;
#pragma unroll
            for (int j = 0; j < 4; ++j)
                Bt[(cg * 4 + j) * BT_STRIDE + kk] = (__bf16)pre[r][j];
        }
    };

    loadTile(0);
    for (int k0 = 0; k0 < D_MODEL; k0 += 32) {
        storeTile();
        __syncthreads();
        if (k0 + 32 < D_MODEL) loadTile(k0 + 32);   // overlaps with WMMAs below
        const v16bf a0 = load_frag16(A0 + k0, halfOff);
        const v16bf a1 = load_frag16(A1 + k0, halfOff);
#pragma unroll
        for (int nt = 0; nt < 4; ++nt) {
            const __bf16* brow = &Bt[(waveN * 64 + nt * 16 + (lane & 15)) * BT_STRIDE];
            const v16bf b = load_frag16(brow, halfOff);
            acc[0][nt] = __builtin_amdgcn_wmma_f32_16x16x32_bf16(
                false, a0, false, b, (short)0, acc[0][nt], false, false);
            acc[1][nt] = __builtin_amdgcn_wmma_f32_16x16x32_bf16(
                false, a1, false, b, (short)0, acc[1][nt], false, false);
        }
        __syncthreads();
    }

    const int mbase = mblk + waveM * 32 + 8 * (lane >> 4);
#pragma unroll
    for (int nt = 0; nt < 4; ++nt) {
        const int fcol = nblk + waveN * 64 + nt * 16 + (lane & 15);
        const float bias = b1[e * D_FF + fcol];
#pragma unroll
        for (int f = 0; f < 2; ++f)
#pragma unroll
            for (int v = 0; v < 8; ++v) {
                const int c = mbase + f * 16 + v;
                const float g = gelu_tanh(acc[f][nt][v] + bias);
                H[((size_t)e * CAP + c) * D_FF + fcol] = (__bf16)g;
            }
    }
}

__global__ __launch_bounds__(256) void moe_ffn2(
        const float* __restrict__ w2, const float* __restrict__ b2,
        const __bf16* __restrict__ H, const int* __restrict__ slot_token,
        const float* __restrict__ gate, float* __restrict__ out) {
    const int e = blockIdx.z;
    const int mblk = blockIdx.x * 128;
    const int nblk = blockIdx.y * 128;
    const int tid = threadIdx.x;
    const int lane = tid & 31, wid = tid >> 5;
    const int waveM = wid & 3, waveN = wid >> 2;
    const int halfOff = (lane >> 4) * 8;

    __shared__ __bf16 Bt[128 * BT_STRIDE];

    const float* W = w2 + (size_t)e * D_FF * D_MODEL;
    const __bf16* A0 = H + ((size_t)e * CAP + mblk + waveM * 32 + (lane & 15)) * D_FF;
    const __bf16* A1 = A0 + (size_t)16 * D_FF;

    v8f acc[2][4];
#pragma unroll
    for (int f = 0; f < 2; ++f)
#pragma unroll
        for (int nt = 0; nt < 4; ++nt) acc[f][nt] = v8f{};

    const int cg = tid & 31;
    const int rg = tid >> 5;
    v4f pre[4];

    auto loadTile = [&](int k0) {
#pragma unroll
        for (int r = 0; r < 4; ++r)
            pre[r] = *(const v4f*)(W + (size_t)(k0 + rg * 4 + r) * D_MODEL + nblk + cg * 4);
    };
    auto storeTile = [&]() {
#pragma unroll
        for (int r = 0; r < 4; ++r) {
            const int kk = rg * 4 + r;
#pragma unroll
            for (int j = 0; j < 4; ++j)
                Bt[(cg * 4 + j) * BT_STRIDE + kk] = (__bf16)pre[r][j];
        }
    };

    loadTile(0);
    for (int k0 = 0; k0 < D_FF; k0 += 32) {
        storeTile();
        __syncthreads();
        if (k0 + 32 < D_FF) loadTile(k0 + 32);
        const v16bf a0 = load_frag16(A0 + k0, halfOff);
        const v16bf a1 = load_frag16(A1 + k0, halfOff);
#pragma unroll
        for (int nt = 0; nt < 4; ++nt) {
            const __bf16* brow = &Bt[(waveN * 64 + nt * 16 + (lane & 15)) * BT_STRIDE];
            const v16bf b = load_frag16(brow, halfOff);
            acc[0][nt] = __builtin_amdgcn_wmma_f32_16x16x32_bf16(
                false, a0, false, b, (short)0, acc[0][nt], false, false);
            acc[1][nt] = __builtin_amdgcn_wmma_f32_16x16x32_bf16(
                false, a1, false, b, (short)0, acc[1][nt], false, false);
        }
        __syncthreads();
    }

    // fused combine: scatter rows back to tokens, scaled by gate.
    float bias[4]; int ncol[4];
#pragma unroll
    for (int nt = 0; nt < 4; ++nt) {
        ncol[nt] = nblk + waveN * 64 + nt * 16 + (lane & 15);
        bias[nt] = b2[e * D_MODEL + ncol[nt]];
    }
    const int mbase = mblk + waveM * 32 + 8 * (lane >> 4);
#pragma unroll
    for (int f = 0; f < 2; ++f) {
        int   stok[8];
        float gval[8];
#pragma unroll
        for (int v = 0; v < 8; ++v) {
            const int c = mbase + f * 16 + v;
            stok[v] = slot_token[e * CAP + c];
            gval[v] = (stok[v] >= 0) ? gate[stok[v]] : 0.0f;
        }
#pragma unroll
        for (int nt = 0; nt < 4; ++nt)
#pragma unroll
            for (int v = 0; v < 8; ++v)
                if (stok[v] >= 0)
                    out[(size_t)stok[v] * D_MODEL + ncol[nt]] =
                        gval[v] * (acc[f][nt][v] + bias[nt]);
    }
}

// ---------------------------------------------------------------------------
// host launcher
// ---------------------------------------------------------------------------
extern "C" void kernel_launch(void* const* d_in, const int* in_sizes, int n_in,
                              void* d_out, int out_size, void* d_ws, size_t ws_size,
                              hipStream_t stream) {
    const float* x  = (const float*)d_in[0];   // [4,2048,1024]
    const float* wg = (const float*)d_in[1];   // [1024,16]
    const float* w1 = (const float*)d_in[2];   // [16,1024,4096]
    const float* b1 = (const float*)d_in[3];   // [16,4096]
    const float* w2 = (const float*)d_in[4];   // [16,4096,1024]
    const float* b2 = (const float*)d_in[5];   // [16,1024]
    float* out = (float*)d_out;                // [S*M] out, [1] l_aux, [16] exp_counts

    char* ws = (char*)d_ws;
    float*  me_sum     = (float*)(ws + 0);          // 16 f32
    int*    expert_id  = (int*)  (ws + 256);        // 8192 i32
    float*  gate       = (float*)(ws + 256 + 32768);
    int*    slot_token = (int*)  (ws + 256 + 65536);
    __bf16* Xd         = (__bf16*)(ws + 98560);                 // 16 MB
    __bf16* H          = (__bf16*)(ws + 98560 + 16777216);      // 64 MB

    const size_t n_main = (size_t)S_TOK * D_MODEL;

    moe_init<<<(unsigned)((n_main + 255) / 256), 256, 0, stream>>>(out, me_sum, slot_token);
    moe_gate<<<S_TOK / 8, 256, 0, stream>>>(x, wg, expert_id, gate, me_sum);
    moe_scan<<<1, 512, 0, stream>>>(expert_id, me_sum, slot_token, out + n_main);
    moe_pack<<<N_EXP * CAP, 256, 0, stream>>>(x, slot_token, Xd);
    moe_ffn1<<<dim3(CAP / 128, D_FF / 128, N_EXP), 256, 0, stream>>>(w1, b1, Xd, H);
    moe_ffn2<<<dim3(CAP / 128, D_MODEL / 128, N_EXP), 256, 0, stream>>>(w2, b2, H, slot_token, gate, out);
}
